// FLH2_10402410791636
// MI455X (gfx1250) — matrix operand order repaired and checked by reading, not compiled, not run on hardware
//
#include <hip/hip_runtime.h>
#include <stdint.h>
#include <stddef.h>

#define B_    32
#define T_    1024
#define D_    384
#define NEXP  1025
#define ALPHA 0.1f

typedef __bf16 bf16;
typedef bf16  v16bf  __attribute__((ext_vector_type(16)));
typedef bf16  bf16x2 __attribute__((ext_vector_type(2)));
typedef float v8f    __attribute__((ext_vector_type(8)));
typedef float f32x4  __attribute__((ext_vector_type(4)));
typedef unsigned int u32x2 __attribute__((ext_vector_type(2)));
typedef unsigned int u32x4 __attribute__((ext_vector_type(4)));

// native f32->bf16 (RTNE); pairs lower to v_cvt_pk_bf16_f32
__device__ __forceinline__ uint32_t pkbf(float a, float b) {
  bf16x2 v;
  v.x = (bf16)a;
  v.y = (bf16)b;
  return __builtin_bit_cast(uint32_t, v);
}

union V16U { v16bf v; u32x4 q[2]; };
__device__ __forceinline__ v16bf ld_bf16x16(const bf16* p0, const bf16* p1) {
  V16U u;
  u.q[0] = *(const u32x4*)p0;   // ds_load_b128
  u.q[1] = *(const u32x4*)p1;   // ds_load_b128
  return u.v;
}

// ---------------------------------------------------------------------------
// Kernel 0: per-(b,t) row stats of x (max, sumexp) for the softmax target.
// ---------------------------------------------------------------------------
__global__ __launch_bounds__(256) void FLH2_stats_kernel(
    const float* __restrict__ x, float* __restrict__ xmax, float* __restrict__ xsum) {
  const int row  = blockIdx.x * 8 + (threadIdx.x >> 5);   // row in [0, B*T)
  const int lane = threadIdx.x & 31;
  const float* xr = x + (size_t)row * D_;
  float m = -3.4e38f;
  for (int d = lane; d < D_; d += 32) m = fmaxf(m, xr[d]);
#pragma unroll
  for (int off = 16; off >= 1; off >>= 1) m = fmaxf(m, __shfl_xor(m, off));
  float s = 0.0f;
  for (int d = lane; d < D_; d += 32) s += __expf(xr[d] - m);
#pragma unroll
  for (int off = 16; off >= 1; off >>= 1) s += __shfl_xor(s, off);
  if (lane == 0) { xmax[row] = m; xsum[row] = s; }
}

// ---------------------------------------------------------------------------
// Kernel 1: per-(t,k) loss.  outs = x_t @ W[e] + b[e]  via bf16 WMMA,
// loss = -(1/B) sum_b [ sum_d softmax(x_t)_bd * outs_bd - lse_b(outs) ].
// 256 threads (8 waves); K-chunk = 64 (two WMMA K-steps per stage).
// ---------------------------------------------------------------------------
__global__ __launch_bounds__(256) void FLH2_loss_kernel(
    const float* __restrict__ x, const float* __restrict__ W,
    const float* __restrict__ bia, const int* __restrict__ aidx,
    const float* __restrict__ amask, const float* __restrict__ xmax,
    const float* __restrict__ xsum, float* __restrict__ loss_out, int K) {
  const int t   = blockIdx.x;
  const int k   = blockIdx.y;
  const int tid = threadIdx.x;
  const float mk = amask[t * K + k];
  if (mk == 0.0f) { if (tid == 0) loss_out[t * K + k] = 0.0f; return; }
  const int e = aidx[t * K + k];

  __shared__ __align__(16) bf16 xb[32][72];    // x chunk  (32 x 64), stride 144B
  __shared__ __align__(16) bf16 wb[384][72];   // W chunk transposed: [n][kk 0..63]
  __shared__ float redw[8][32];
  __shared__ float rowmax[32];
  __shared__ float rowlse[32];

  const int wid = tid >> 5, lane = tid & 31, ln = lane & 15, hh = lane >> 4;

  v8f acc[2][3];
#pragma unroll
  for (int mt = 0; mt < 2; ++mt)
#pragma unroll
    for (int j = 0; j < 3; ++j)
#pragma unroll
      for (int r = 0; r < 8; ++r) acc[mt][j][r] = 0.0f;

  const int srow = tid >> 3, sc0 = (tid & 7) << 3;          // 32 rows x 8 cols
  const size_t xrowbase = ((size_t)srow * T_ + t) * D_;
  const size_t wbase    = (size_t)e * ((size_t)D_ * D_);

  for (int kc = 0; kc < 6; ++kc) {                           // 6 stages of K=64
    // stage x chunk: 32x64, each thread 8 cols -> one b128 store
    {
      const float* xp = x + xrowbase + kc * 64 + sc0;
      f32x4 x0 = *(const f32x4*)(xp);
      f32x4 x1 = *(const f32x4*)(xp + 4);
      u32x4 pk;
      pk.x = pkbf(x0.x, x0.y);  pk.y = pkbf(x0.z, x0.w);
      pk.z = pkbf(x1.x, x1.y);  pk.w = pkbf(x1.z, x1.w);
      *(u32x4*)&xb[srow][sc0] = pk;
    }
    // stage W chunk (64 x 384) transposed, 4x4 micro-tiles -> b64 stores
#pragma unroll
    for (int it = 0; it < 6; ++it) {
      int q  = tid + it * 256;              // 0..1535
      int rg = q / 96, cg = q % 96;
      int r0 = rg * 4, c0 = cg * 4;
      const float* wp = W + wbase + (size_t)(kc * 64 + r0) * D_ + c0;
      f32x4 w0 = *(const f32x4*)(wp);
      f32x4 w1 = *(const f32x4*)(wp + D_);
      f32x4 w2 = *(const f32x4*)(wp + 2 * D_);
      f32x4 w3 = *(const f32x4*)(wp + 3 * D_);
      u32x2 p0; p0.x = pkbf(w0.x, w1.x); p0.y = pkbf(w2.x, w3.x);
      u32x2 p1; p1.x = pkbf(w0.y, w1.y); p1.y = pkbf(w2.y, w3.y);
      u32x2 p2; p2.x = pkbf(w0.z, w1.z); p2.y = pkbf(w2.z, w3.z);
      u32x2 p3; p3.x = pkbf(w0.w, w1.w); p3.y = pkbf(w2.w, w3.w);
      *(u32x2*)&wb[c0 + 0][r0] = p0;
      *(u32x2*)&wb[c0 + 1][r0] = p1;
      *(u32x2*)&wb[c0 + 2][r0] = p2;
      *(u32x2*)&wb[c0 + 3][r0] = p3;
    }
    if (kc + 1 < 6)
      __builtin_prefetch(W + wbase + (size_t)((kc + 1) * 64 + (tid >> 2)) * D_, 0, 0);
    __syncthreads();

    v16bf a0  = ld_bf16x16(&xb[ln][8 * hh],           &xb[ln][16 + 8 * hh]);
    v16bf a1  = ld_bf16x16(&xb[16 + ln][8 * hh],      &xb[16 + ln][16 + 8 * hh]);
    v16bf a0b = ld_bf16x16(&xb[ln][32 + 8 * hh],      &xb[ln][48 + 8 * hh]);
    v16bf a1b = ld_bf16x16(&xb[16 + ln][32 + 8 * hh], &xb[16 + ln][48 + 8 * hh]);
#pragma unroll
    for (int j = 0; j < 3; ++j) {
      int n = (wid * 3 + j) * 16 + ln;
      v16bf b0 = ld_bf16x16(&wb[n][8 * hh],      &wb[n][16 + 8 * hh]);
      v16bf b1 = ld_bf16x16(&wb[n][32 + 8 * hh], &wb[n][48 + 8 * hh]);
      acc[0][j] = __builtin_amdgcn_wmma_f32_16x16x32_bf16(false, a0,  false, b0,
                                                          (short)0, acc[0][j], false, false);
      acc[1][j] = __builtin_amdgcn_wmma_f32_16x16x32_bf16(false, a1,  false, b0,
                                                          (short)0, acc[1][j], false, false);
      acc[0][j] = __builtin_amdgcn_wmma_f32_16x16x32_bf16(false, a0b, false, b1,
                                                          (short)0, acc[0][j], false, false);
      acc[1][j] = __builtin_amdgcn_wmma_f32_16x16x32_bf16(false, a1b, false, b1,
                                                          (short)0, acc[1][j], false, false);
    }
    __syncthreads();
  }

  // add bias
  float bcol[3];
#pragma unroll
  for (int j = 0; j < 3; ++j) bcol[j] = bia[(size_t)e * D_ + (wid * 3 + j) * 16 + ln];
#pragma unroll
  for (int mt = 0; mt < 2; ++mt)
#pragma unroll
    for (int j = 0; j < 3; ++j)
#pragma unroll
      for (int r = 0; r < 8; ++r) acc[mt][j][r] += bcol[j];

  // phase 1: row max
#pragma unroll
  for (int mt = 0; mt < 2; ++mt)
#pragma unroll
    for (int r = 0; r < 8; ++r) {
      float v = fmaxf(acc[mt][0][r], fmaxf(acc[mt][1][r], acc[mt][2][r]));
#pragma unroll
      for (int off = 8; off >= 1; off >>= 1) v = fmaxf(v, __shfl_xor(v, off));
      if (ln == 0) redw[wid][16 * mt + r + 8 * hh] = v;
    }
  __syncthreads();
  if (tid < 32) {
    float v = redw[0][tid];
#pragma unroll
    for (int w2 = 1; w2 < 8; ++w2) v = fmaxf(v, redw[w2][tid]);
    rowmax[tid] = v;
  }
  __syncthreads();
  // phase 2: lse
#pragma unroll
  for (int mt = 0; mt < 2; ++mt)
#pragma unroll
    for (int r = 0; r < 8; ++r) {
      int M = 16 * mt + r + 8 * hh;
      float rm = rowmax[M];
      float s = __expf(acc[mt][0][r] - rm) + __expf(acc[mt][1][r] - rm) +
                __expf(acc[mt][2][r] - rm);
#pragma unroll
      for (int off = 8; off >= 1; off >>= 1) s += __shfl_xor(s, off);
      if (ln == 0) redw[wid][M] = s;
    }
  __syncthreads();
  if (tid < 32) {
    float s = 0.0f;
#pragma unroll
    for (int w2 = 0; w2 < 8; ++w2) s += redw[w2][tid];
    rowlse[tid] = rowmax[tid] + __logf(s);
  }
  __syncthreads();
  // phase 3: target dot
#pragma unroll
  for (int mt = 0; mt < 2; ++mt)
#pragma unroll
    for (int r = 0; r < 8; ++r) {
      int M = 16 * mt + r + 8 * hh;
      float xm = xmax[(size_t)M * T_ + t];
      float xs = xsum[(size_t)M * T_ + t];
      float d = 0.0f;
#pragma unroll
      for (int j = 0; j < 3; ++j) {
        int n = (wid * 3 + j) * 16 + ln;
        float xv = x[((size_t)M * T_ + t) * D_ + n];
        d += (__expf(xv - xm) / xs) * acc[mt][j][r];
      }
#pragma unroll
      for (int off = 8; off >= 1; off >>= 1) d += __shfl_xor(d, off);
      if (ln == 0) redw[wid][M] = d;
    }
  __syncthreads();
  if (tid < 32) {
    float d = 0.0f;
#pragma unroll
    for (int w2 = 0; w2 < 8; ++w2) d += redw[w2][tid];
    float v = d - rowlse[tid];
#pragma unroll
    for (int off = 16; off >= 1; off >>= 1) v += __shfl_xor(v, off);
    if (tid == 0) loss_out[t * K + k] = -v / (float)B_;
  }
}

// ---------------------------------------------------------------------------
// Kernel 2: sequential multiplicative-weights scan (cheap, 1 block).
// ---------------------------------------------------------------------------
__global__ __launch_bounds__(64) void FLH2_scan_kernel(
    const float* __restrict__ loss, const int* __restrict__ aidx,
    const float* __restrict__ amask, float* __restrict__ wout, int K) {
  __shared__ float p[NEXP];
  __shared__ float red2[2];
  const int tid = threadIdx.x;
  for (int i = tid; i < NEXP; i += 64) p[i] = 0.0f;
  __syncthreads();
  if (tid == 0) p[0] = 1.0f;
  __syncthreads();
  for (int t = 0; t < T_; ++t) {
    int e = 0; float mk = 0.0f, pa = 0.0f;
    if (tid < K) {
      e  = aidx[t * K + tid];
      mk = amask[t * K + tid];
      pa = p[e] * mk;
    }
    float s = pa;
#pragma unroll
    for (int off = 16; off >= 1; off >>= 1) s += __shfl_xor(s, off);
    if ((tid & 31) == 0) red2[tid >> 5] = s;
    __syncthreads();
    s = red2[0] + red2[1];
    if (tid < K) wout[t * K + tid] = (s > 0.0f) ? pa / s : 0.0f;
    float pu = 0.0f;
    if (tid < K) pu = pa * __expf(-ALPHA * loss[t * K + tid]) * mk;
    float s2 = pu;
#pragma unroll
    for (int off = 16; off >= 1; off >>= 1) s2 += __shfl_xor(s2, off);
    __syncthreads();
    if ((tid & 31) == 0) red2[tid >> 5] = s2;
    __syncthreads();
    s2 = red2[0] + red2[1];
    if (s2 > 0.0f) pu /= s2;
    if (tid < K && mk > 0.0f) p[e] = pu;
    if (tid == 0) p[t + 1] = 1.0f / (float)(t + 1);
    __syncthreads();
  }
}

// ---------------------------------------------------------------------------
// Kernel 3: y_t = x_t @ (sum_k w_k W[e_k]) + sum_k w_k b[e_k].
// w_k folded into B staging; one WMMA accumulator across all alive experts.
// ---------------------------------------------------------------------------
__global__ __launch_bounds__(256) void FLH2_y_kernel(
    const float* __restrict__ x, const float* __restrict__ W,
    const float* __restrict__ bia, const int* __restrict__ aidx,
    const float* __restrict__ wts, float* __restrict__ out, int K) {
  const int t   = blockIdx.x;
  const int tid = threadIdx.x;
  __shared__ __align__(16) bf16 xball[32][392];  // full x_t bf16, stride 784B
  __shared__ __align__(16) bf16 wb[384][40];     // K-chunk 32, transposed
  __shared__ float wloc[64];
  __shared__ int   eloc[64];
  __shared__ float bbar[384];

  if (tid < K) { wloc[tid] = wts[t * K + tid]; eloc[tid] = aidx[t * K + tid]; }
#pragma unroll
  for (int it = 0; it < 12; ++it) {
    int lin = tid + it * 256;                 // f32x4 index, 3072 total
    int r = lin / 96, c0 = (lin % 96) * 4;
    f32x4 xv = *(const f32x4*)(x + ((size_t)r * T_ + t) * D_ + c0);
    u32x2 pk; pk.x = pkbf(xv.x, xv.y); pk.y = pkbf(xv.z, xv.w);
    *(u32x2*)&xball[r][c0] = pk;
  }
  __syncthreads();
  for (int d = tid; d < D_; d += 256) {
    float s = 0.0f;
    for (int k2 = 0; k2 < K; ++k2) s += wloc[k2] * bia[(size_t)eloc[k2] * D_ + d];
    bbar[d] = s;
  }

  const int wid = tid >> 5, lane = tid & 31, ln = lane & 15, hh = lane >> 4;
  v8f acc[2][3];
#pragma unroll
  for (int mt = 0; mt < 2; ++mt)
#pragma unroll
    for (int j = 0; j < 3; ++j)
#pragma unroll
      for (int r = 0; r < 8; ++r) acc[mt][j][r] = 0.0f;

  for (int k2 = 0; k2 < K; ++k2) {
    float wk = wloc[k2];
    if (wk == 0.0f) continue;                 // uniform across block
    const size_t wbase = (size_t)eloc[k2] * ((size_t)D_ * D_);
    for (int kc = 0; kc < 12; ++kc) {
      // stage scaled W chunk (32 x 384) transposed, 4x4 micro-tiles
#pragma unroll
      for (int it = 0; it < 3; ++it) {
        int q  = tid + it * 256;              // 0..767
        int rg = q / 96, cg = q % 96;
        int r0 = rg * 4, c0 = cg * 4;
        const float* wp = W + wbase + (size_t)(kc * 32 + r0) * D_ + c0;
        f32x4 w0 = *(const f32x4*)(wp);
        f32x4 w1 = *(const f32x4*)(wp + D_);
        f32x4 w2 = *(const f32x4*)(wp + 2 * D_);
        f32x4 w3 = *(const f32x4*)(wp + 3 * D_);
        u32x2 p0; p0.x = pkbf(wk * w0.x, wk * w1.x); p0.y = pkbf(wk * w2.x, wk * w3.x);
        u32x2 p1; p1.x = pkbf(wk * w0.y, wk * w1.y); p1.y = pkbf(wk * w2.y, wk * w3.y);
        u32x2 p2; p2.x = pkbf(wk * w0.z, wk * w1.z); p2.y = pkbf(wk * w2.z, wk * w3.z);
        u32x2 p3; p3.x = pkbf(wk * w0.w, wk * w1.w); p3.y = pkbf(wk * w2.w, wk * w3.w);
        *(u32x2*)&wb[c0 + 0][r0] = p0;
        *(u32x2*)&wb[c0 + 1][r0] = p1;
        *(u32x2*)&wb[c0 + 2][r0] = p2;
        *(u32x2*)&wb[c0 + 3][r0] = p3;
      }
      __syncthreads();
      v16bf a0 = ld_bf16x16(&xball[ln][kc * 32 + 8 * hh],
                            &xball[ln][kc * 32 + 16 + 8 * hh]);
      v16bf a1 = ld_bf16x16(&xball[16 + ln][kc * 32 + 8 * hh],
                            &xball[16 + ln][kc * 32 + 16 + 8 * hh]);
#pragma unroll
      for (int j = 0; j < 3; ++j) {
        int n = (wid * 3 + j) * 16 + ln;
        v16bf bop = ld_bf16x16(&wb[n][8 * hh], &wb[n][16 + 8 * hh]);
        acc[0][j] = __builtin_amdgcn_wmma_f32_16x16x32_bf16(false, a0, false, bop,
                                                            (short)0, acc[0][j], false, false);
        acc[1][j] = __builtin_amdgcn_wmma_f32_16x16x32_bf16(false, a1, false, bop,
                                                            (short)0, acc[1][j], false, false);
      }
      __syncthreads();
    }
  }
  __syncthreads();
#pragma unroll
  for (int mt = 0; mt < 2; ++mt)
#pragma unroll
    for (int j = 0; j < 3; ++j) {
      int n = (wid * 3 + j) * 16 + ln;
      float bb = bbar[n];
#pragma unroll
      for (int r = 0; r < 8; ++r) {
        int mrow = 16 * mt + r + 8 * hh;
        out[((size_t)mrow * T_ + t) * D_ + n] = acc[mt][j][r] + bb;
      }
    }
}

// ---------------------------------------------------------------------------
extern "C" void kernel_launch(void* const* d_in, const int* in_sizes, int n_in,
                              void* d_out, int out_size, void* d_ws, size_t ws_size,
                              hipStream_t stream) {
  const float* x  = (const float*)d_in[0];
  const float* W  = (const float*)d_in[1];
  const float* bi = (const float*)d_in[2];
  const int*   ai = (const int*)d_in[3];
  const float* am = (const float*)d_in[4];
  float* out = (float*)d_out;
  const int K = in_sizes[3] / T_;

  float* xmaxb = (float*)d_ws;
  float* xsumb = xmaxb + (size_t)B_ * T_;
  float* lossb = xsumb + (size_t)B_ * T_;
  float* wbuf  = lossb + (size_t)T_ * K;

  FLH2_stats_kernel<<<dim3((B_ * T_) / 8), 256, 0, stream>>>(x, xmaxb, xsumb);
  FLH2_loss_kernel<<<dim3(T_, K), 256, 0, stream>>>(x, W, bi, ai, am, xmaxb, xsumb, lossb, K);
  FLH2_scan_kernel<<<1, 64, 0, stream>>>(lossb, ai, am, wbuf, K);
  FLH2_y_kernel<<<dim3(T_), 256, 0, stream>>>(x, W, bi, ai, wbuf, out, K);
}
